// vLSTMCell_39350490366667
// MI455X (gfx1250) — matrix operand order repaired and checked by reading, not compiled, not run on hardware
//
#include <hip/hip_runtime.h>
#include <math.h>

typedef __attribute__((ext_vector_type(16))) __bf16 v16bf;
typedef __attribute__((ext_vector_type(8)))  __bf16 v8bf;
typedef __attribute__((ext_vector_type(4)))  __bf16 v4bf;
typedef __attribute__((ext_vector_type(8)))  float  v8f;

#define B_DIM  8192
#define IN_DIM 2048
#define H_DIM  2048
#define FOURH  8192
#define KTOT   4096

#define M_TILE 128
#define N_TILE 64
#define K_STEP 32
#define NKB    (KTOT / K_STEP)

__device__ __forceinline__ float sigmoidf_(float x) {
    return 1.0f / (1.0f + __expf(-x));
}

// ---- CDNA5 async global->LDS (ASYNCcnt-tracked), via inline asm ----
__device__ __forceinline__ void async_ld_b128(unsigned int lds_off, unsigned long long gaddr) {
    asm volatile("global_load_async_to_lds_b128 %0, %1, off"
                 :: "v"(lds_off), "v"(gaddr) : "memory");
}
__device__ __forceinline__ void wait_async0() {
#if __has_builtin(__builtin_amdgcn_s_wait_asynccnt)
    __builtin_amdgcn_s_wait_asynccnt(0);
#else
    asm volatile("s_wait_asynccnt 0x0" ::: "memory");
#endif
}
__device__ __forceinline__ unsigned int lds_off_u32(const void* p) {
    return (unsigned int)(uintptr_t)p;   // low 32 bits of flat addr = LDS offset
}

// =====================================================================
// Kernel 1: At[row][k] (bf16, [8192][4096]) = concat(inp, h0) row-major
// =====================================================================
__global__ __launch_bounds__(256)
void convert_a_kernel(const float* __restrict__ inp, const float* __restrict__ h0,
                      __bf16* __restrict__ At)
{
    const int tid = threadIdx.x;
    #pragma unroll
    for (int j = 0; j < 4; ++j) {
        const int id  = blockIdx.x * 1024 + j * 256 + tid;   // float4 index
        const int row = id >> 10;                            // 1024 float4 per row
        const int kq  = (id & 1023) << 2;                    // k = 0..4092
        const float* src = (kq < IN_DIM) ? (inp + (size_t)row * IN_DIM + kq)
                                         : (h0  + (size_t)row * IN_DIM + (kq - IN_DIM));
        const float4 v = *(const float4*)src;
        v4bf o; o.x = (__bf16)v.x; o.y = (__bf16)v.y; o.z = (__bf16)v.z; o.w = (__bf16)v.w;
        *(v4bf*)(At + (size_t)row * KTOT + kq) = o;
    }
}

// =====================================================================
// Kernel 2: Vt[col][k] (bf16, [8192][4096]) = transpose of [W ; U]
//           Vt[col][k] = (k < 2048) ? W[k][col] : U[k-2048][col]
// =====================================================================
__global__ __launch_bounds__(256)
void transpose_b_kernel(const float* __restrict__ W, const float* __restrict__ U,
                        __bf16* __restrict__ Vt)
{
    __shared__ float tile[32][33];
    const int tid  = threadIdx.x;
    const int col0 = blockIdx.x * 32;
    const int k0   = blockIdx.y * 32;
    const float* src  = (k0 < IN_DIM) ? W : U;
    const int    kOff = (k0 < IN_DIM) ? k0 : (k0 - IN_DIM);

    #pragma unroll
    for (int j = 0; j < 4; ++j) {
        const int kr = (tid >> 5) + j * 8;   // 0..31
        const int c  = tid & 31;
        tile[kr][c] = src[(size_t)(kOff + kr) * FOURH + col0 + c];
    }
    __syncthreads();
    #pragma unroll
    for (int j = 0; j < 4; ++j) {
        const int cr = (tid >> 5) + j * 8;   // 0..31
        const int k  = tid & 31;
        Vt[(size_t)(col0 + cr) * KTOT + k0 + k] = (__bf16)tile[k][cr];
    }
}

// =====================================================================
// Kernel 3: async double-buffered bf16 WMMA GEMM + fused LSTM epilogue
// =====================================================================
__global__ __launch_bounds__(256)
void lstm_wmma_async_kernel(const __bf16* __restrict__ At, const __bf16* __restrict__ Vt,
                            const float* __restrict__ c0,
                            const float* __restrict__ b_w, const float* __restrict__ b_u,
                            float* __restrict__ out)
{
    __shared__ __align__(16) __bf16 bufA[2][M_TILE][K_STEP];          // 2 x 8 KB
    __shared__ __align__(16) __bf16 bufB[2][4][N_TILE][K_STEP];       // 2 x 16 KB

    const int tid  = threadIdx.x;
    const int wave = tid >> 5;
    const int lane = tid & 31;
    const int half = lane >> 4;
    const int l15  = lane & 15;
    const int wm   = wave >> 1;     // 0..3
    const int wn   = wave & 1;      // 0..1

    const int m0 = blockIdx.y * M_TILE;
    const int n0 = blockIdx.x * N_TILE;

    // --- per-thread static staging addresses -------------------------------
    // A: 512 16B-chunks per tile; this thread owns chunks tid, tid+256.
    // chunk c: row = c>>2 (0..127), part = c&3 (16B = 8 bf16 along k)
    unsigned long long gA[2];  unsigned int lA[2];
    #pragma unroll
    for (int j = 0; j < 2; ++j) {
        const int c    = tid + j * 256;
        const int row  = c >> 2;
        const int part = c & 3;
        gA[j] = (unsigned long long)(uintptr_t)(At + (size_t)(m0 + row) * KTOT + part * 8);
        lA[j] = lds_off_u32(&bufA[0][row][part * 8]);
    }
    // B: 1024 16B-chunks per tile; this thread owns 4.
    // chunk c: cc = c>>2 -> gate g = cc>>6, col n = cc&63 ; part = c&3
    unsigned long long gB[4];  unsigned int lB[4];
    #pragma unroll
    for (int j = 0; j < 4; ++j) {
        const int c    = tid + j * 256;
        const int cc   = c >> 2;
        const int g    = cc >> 6;
        const int n    = cc & 63;
        const int part = c & 3;
        gB[j] = (unsigned long long)(uintptr_t)(Vt + (size_t)(g * H_DIM + n0 + n) * KTOT + part * 8);
        lB[j] = lds_off_u32(&bufB[0][g][n][part * 8]);
    }

    v8f acc[4][2][2];
    const v8f vzero = {0.f, 0.f, 0.f, 0.f, 0.f, 0.f, 0.f, 0.f};
    #pragma unroll
    for (int g = 0; g < 4; ++g)
        #pragma unroll
        for (int mi = 0; mi < 2; ++mi)
            #pragma unroll
            for (int ni = 0; ni < 2; ++ni)
                acc[g][mi][ni] = vzero;

    // --- prologue: stage tile 0 into buffer 0 ------------------------------
    {
        #pragma unroll
        for (int j = 0; j < 2; ++j) async_ld_b128(lA[j], gA[j]);
        #pragma unroll
        for (int j = 0; j < 4; ++j) async_ld_b128(lB[j], gB[j]);
    }
    wait_async0();
    __syncthreads();

    for (int kb = 0; kb < NKB; ++kb) {
        const int p = kb & 1;

        // stage tile kb+1 into the other buffer while we compute on p
        if (kb + 1 < NKB) {
            const unsigned long long gshift = (unsigned long long)(kb + 1) * (K_STEP * 2);
            const unsigned int lshiftA = (unsigned int)((1 - p) * sizeof(bufA[0]));
            const unsigned int lshiftB = (unsigned int)((1 - p) * sizeof(bufB[0]));
            #pragma unroll
            for (int j = 0; j < 2; ++j) async_ld_b128(lA[j] + lshiftA, gA[j] + gshift);
            #pragma unroll
            for (int j = 0; j < 4; ++j) async_ld_b128(lB[j] + lshiftB, gB[j] + gshift);
        }

        // ---- compute on buffer p ----
        v16bf afrag[2];
        #pragma unroll
        for (int mi = 0; mi < 2; ++mi) {
            const __bf16* pa = &bufA[p][wm * 32 + mi * 16 + l15][half * 8];
            const v8bf lo = *(const v8bf*)pa;
            const v8bf hi = *(const v8bf*)(pa + 16);
            afrag[mi] = __builtin_shufflevector(lo, hi,
                0, 1, 2, 3, 4, 5, 6, 7, 8, 9, 10, 11, 12, 13, 14, 15);
        }
        #pragma unroll
        for (int g = 0; g < 4; ++g) {
            #pragma unroll
            for (int ni = 0; ni < 2; ++ni) {
                const __bf16* pb = &bufB[p][g][wn * 32 + ni * 16 + l15][half * 8];
                const v8bf lo = *(const v8bf*)pb;
                const v8bf hi = *(const v8bf*)(pb + 16);
                const v16bf bfrag = __builtin_shufflevector(lo, hi,
                    0, 1, 2, 3, 4, 5, 6, 7, 8, 9, 10, 11, 12, 13, 14, 15);
                #pragma unroll
                for (int mi = 0; mi < 2; ++mi) {
                    acc[g][mi][ni] = __builtin_amdgcn_wmma_f32_16x16x32_bf16(
                        false, afrag[mi], false, bfrag,
                        (short)0, acc[g][mi][ni], false, false);
                }
            }
        }

        wait_async0();       // our async writes into the other buffer are done
        __syncthreads();     // everyone's writes visible; prior reads complete
    }

    // ---- fused epilogue: gates -> (h, c) ----
    const int rowBase = m0 + wm * 32;
    #pragma unroll
    for (int mi = 0; mi < 2; ++mi) {
        #pragma unroll
        for (int ni = 0; ni < 2; ++ni) {
            const int gcol = n0 + wn * 32 + ni * 16 + l15;
            const float bi = b_w[0 * H_DIM + gcol] + b_u[0 * H_DIM + gcol];
            const float bf = b_w[1 * H_DIM + gcol] + b_u[1 * H_DIM + gcol];
            const float bo = b_w[2 * H_DIM + gcol] + b_u[2 * H_DIM + gcol];
            const float bg = b_w[3 * H_DIM + gcol] + b_u[3 * H_DIM + gcol];
            #pragma unroll
            for (int e = 0; e < 8; ++e) {
                const int row = rowBase + mi * 16 + e + 8 * half;
                const float ig = sigmoidf_(acc[0][mi][ni][e] + bi);
                const float fg = sigmoidf_(acc[1][mi][ni][e] + bf);
                const float og = sigmoidf_(acc[2][mi][ni][e] + bo);
                const float gg = tanhf(acc[3][mi][ni][e] + bg);
                const float cc = fg * c0[(size_t)row * H_DIM + gcol] + ig * gg;
                const float hh = og * tanhf(cc);
                out[(size_t)row * H_DIM + gcol] = hh;
                out[(size_t)B_DIM * H_DIM + (size_t)row * H_DIM + gcol] = cc;
            }
        }
    }
}

// =====================================================================
// Fallback (no workspace): round-1 fused kernel with in-loop conversion
// =====================================================================
__global__ __launch_bounds__(256)
void lstm_wmma_kernel(const float* __restrict__ inp, const float* __restrict__ h0,
                      const float* __restrict__ c0,  const float* __restrict__ W,
                      const float* __restrict__ b_w, const float* __restrict__ U,
                      const float* __restrict__ b_u, float* __restrict__ out)
{
    __shared__ __align__(16) __bf16 lds_a[M_TILE][K_STEP];
    __shared__ __align__(16) __bf16 lds_b[4][N_TILE][K_STEP];

    const int tid  = threadIdx.x;
    const int wave = tid >> 5;
    const int lane = tid & 31;
    const int half = lane >> 4;
    const int l15  = lane & 15;
    const int wm   = wave >> 1;
    const int wn   = wave & 1;
    const int m0 = blockIdx.y * M_TILE;
    const int n0 = blockIdx.x * N_TILE;

    v8f acc[4][2][2];
    const v8f vzero = {0.f, 0.f, 0.f, 0.f, 0.f, 0.f, 0.f, 0.f};
    #pragma unroll
    for (int g = 0; g < 4; ++g)
        #pragma unroll
        for (int mi = 0; mi < 2; ++mi)
            #pragma unroll
            for (int ni = 0; ni < 2; ++ni)
                acc[g][mi][ni] = vzero;

    for (int kb = 0; kb < NKB; ++kb) {
        const int kg = kb * K_STEP;
        const bool fh = (kg < IN_DIM);
        const float* aSrc = fh ? inp : h0;
        const float* bSrc = fh ? W   : U;
        const int kOff = fh ? kg : (kg - IN_DIM);

        __syncthreads();
        #pragma unroll
        for (int i = 0; i < 4; ++i) {
            const int idx = tid + i * 256;
            const int row = idx >> 3;
            const int kq  = (idx & 7) << 2;
            const float4 v = *(const float4*)(aSrc + (size_t)(m0 + row) * IN_DIM + kOff + kq);
            v4bf o; o.x = (__bf16)v.x; o.y = (__bf16)v.y; o.z = (__bf16)v.z; o.w = (__bf16)v.w;
            *(v4bf*)&lds_a[row][kq] = o;
        }
        #pragma unroll
        for (int i = 0; i < 8; ++i) {
            const int idx = tid + i * 256;
            const int g   = idx >> 9;
            const int rem = idx & 511;
            const int kr  = rem >> 4;
            const int nq  = (rem & 15) << 2;
            const float4 v = *(const float4*)(bSrc + (size_t)(kOff + kr) * FOURH
                                              + g * H_DIM + n0 + nq);
            lds_b[g][nq + 0][kr] = (__bf16)v.x;
            lds_b[g][nq + 1][kr] = (__bf16)v.y;
            lds_b[g][nq + 2][kr] = (__bf16)v.z;
            lds_b[g][nq + 3][kr] = (__bf16)v.w;
        }
        __syncthreads();

        v16bf afrag[2];
        #pragma unroll
        for (int mi = 0; mi < 2; ++mi) {
            const __bf16* p = &lds_a[wm * 32 + mi * 16 + l15][half * 8];
            const v8bf lo = *(const v8bf*)p;
            const v8bf hi = *(const v8bf*)(p + 16);
            afrag[mi] = __builtin_shufflevector(lo, hi,
                0, 1, 2, 3, 4, 5, 6, 7, 8, 9, 10, 11, 12, 13, 14, 15);
        }
        #pragma unroll
        for (int g = 0; g < 4; ++g) {
            #pragma unroll
            for (int ni = 0; ni < 2; ++ni) {
                const __bf16* p = &lds_b[g][wn * 32 + ni * 16 + l15][half * 8];
                const v8bf lo = *(const v8bf*)p;
                const v8bf hi = *(const v8bf*)(p + 16);
                const v16bf bfrag = __builtin_shufflevector(lo, hi,
                    0, 1, 2, 3, 4, 5, 6, 7, 8, 9, 10, 11, 12, 13, 14, 15);
                #pragma unroll
                for (int mi = 0; mi < 2; ++mi) {
                    acc[g][mi][ni] = __builtin_amdgcn_wmma_f32_16x16x32_bf16(
                        false, afrag[mi], false, bfrag,
                        (short)0, acc[g][mi][ni], false, false);
                }
            }
        }
    }

    const int rowBase = m0 + wm * 32;
    #pragma unroll
    for (int mi = 0; mi < 2; ++mi) {
        #pragma unroll
        for (int ni = 0; ni < 2; ++ni) {
            const int gcol = n0 + wn * 32 + ni * 16 + l15;
            const float bi = b_w[0 * H_DIM + gcol] + b_u[0 * H_DIM + gcol];
            const float bf = b_w[1 * H_DIM + gcol] + b_u[1 * H_DIM + gcol];
            const float bo = b_w[2 * H_DIM + gcol] + b_u[2 * H_DIM + gcol];
            const float bg = b_w[3 * H_DIM + gcol] + b_u[3 * H_DIM + gcol];
            #pragma unroll
            for (int e = 0; e < 8; ++e) {
                const int row = rowBase + mi * 16 + e + 8 * half;
                const float ig = sigmoidf_(acc[0][mi][ni][e] + bi);
                const float fg = sigmoidf_(acc[1][mi][ni][e] + bf);
                const float og = sigmoidf_(acc[2][mi][ni][e] + bo);
                const float gg = tanhf(acc[3][mi][ni][e] + bg);
                const float cc = fg * c0[(size_t)row * H_DIM + gcol] + ig * gg;
                const float hh = og * tanhf(cc);
                out[(size_t)row * H_DIM + gcol] = hh;
                out[(size_t)B_DIM * H_DIM + (size_t)row * H_DIM + gcol] = cc;
            }
        }
    }
}

extern "C" void kernel_launch(void* const* d_in, const int* in_sizes, int n_in,
                              void* d_out, int out_size, void* d_ws, size_t ws_size,
                              hipStream_t stream) {
    (void)in_sizes; (void)n_in; (void)out_size;
    const float* inp = (const float*)d_in[0];
    const float* h0  = (const float*)d_in[1];
    const float* c0  = (const float*)d_in[2];
    const float* W   = (const float*)d_in[3];
    const float* b_w = (const float*)d_in[4];
    const float* U   = (const float*)d_in[5];
    const float* b_u = (const float*)d_in[6];
    float* out = (float*)d_out;

    const size_t atBytes = (size_t)B_DIM * KTOT * sizeof(__bf16);   // 64 MB
    const size_t vtBytes = (size_t)FOURH * KTOT * sizeof(__bf16);   // 64 MB

    if (ws_size >= atBytes + vtBytes) {
        __bf16* At = (__bf16*)d_ws;
        __bf16* Vt = (__bf16*)((char*)d_ws + atBytes);
        convert_a_kernel<<<dim3(B_DIM), 256, 0, stream>>>(inp, h0, At);
        transpose_b_kernel<<<dim3(FOURH / 32, KTOT / 32), 256, 0, stream>>>(W, U, Vt);
        lstm_wmma_async_kernel<<<dim3(H_DIM / N_TILE, B_DIM / M_TILE), 256, 0, stream>>>(
            At, Vt, c0, b_w, b_u, out);
    } else {
        lstm_wmma_kernel<<<dim3(H_DIM / N_TILE, B_DIM / M_TILE), 256, 0, stream>>>(
            inp, h0, c0, W, b_w, U, b_u, out);
    }
}
